// LocalAttention_5669356834644
// MI455X (gfx1250) — matrix operand order repaired and checked by reading, not compiled, not run on hardware
//
#include <hip/hip_runtime.h>

// ---------------------------------------------------------------------------
// MI455X (gfx1250) local-attention.
// Part 1: x1/x2/x3 = w{1,2,3} @ x   -> bf16 WMMA GEMM, LDS-staged activations
//         via global_load_async_to_lds_b32 (ASYNCcnt), double-buffered.
// Part 2: banded-WMMA similarity + per-pixel softmax + coalesced VALU
//         weighted aggregation.
// ---------------------------------------------------------------------------

typedef __attribute__((ext_vector_type(16))) __bf16       v16bf;
typedef __attribute__((ext_vector_type(8)))  float        v8f;
typedef __attribute__((ext_vector_type(4)))  float        v4f;
typedef __attribute__((ext_vector_type(4)))  unsigned int u32x4;

#define BATCH 4
#define CCH   256
#define HH    64
#define WW    64
#define HWPIX (HH * WW)   // 4096
#define KH    7
#define KW    7
#define KK    49
#define XPAD  36          // LDS row stride (floats): 16B-aligned, conflict-floor

__device__ __forceinline__ __bf16 tobf(float f) { return (__bf16)f; }

union BFrag { v16bf v; u32x4 q[2]; };

// ---------------------------------------------------------------------------
// Kernel 1: channel mixing. Grid = 3 (weight matrix) x 4 (batch) x 64 (pixel
// row of 64). Block = 256 threads (8 waves). Per block: M = 256 out-channels,
// N = 64 pixels, K = 256 in 8 steps of 32.
//   - K-slab of x (32ch x 64px, f32) is async-copied to LDS, transposed to
//     [pixel][channel(+pad)], double-buffered; copy overlaps WMMA.
//   - wave wv owns m-tiles {2wv, 2wv+1} and all 4 n-tiles:
//     8 x v_wmma_f32_16x16x32_bf16 per K-step, A reused over n, B over m.
// Outputs channels-last: x1s/x2s bf16 [B,H,W,C], x3s f32 [B,H,W,C].
// ---------------------------------------------------------------------------
__global__ __launch_bounds__(256) void mix_gemm_kernel(
    const float* __restrict__ x,
    const float* __restrict__ w1,
    const float* __restrict__ w2,
    const float* __restrict__ w3,
    __bf16* __restrict__ x1s,
    __bf16* __restrict__ x2s,
    float*  __restrict__ x3s)
{
  __shared__ __align__(16) float xbuf[2][64][XPAD]; // [buf][pixel][channel]

  const int tid  = threadIdx.x;
  const int lane = tid & 31;
  const int wv   = tid >> 5;

  const int slice = blockIdx.x >> 8;         // 0..2 : which weight matrix
  const int b     = (blockIdx.x >> 6) & 3;   // batch
  const int prow  = blockIdx.x & 63;         // image row  (64 pixels = row)
  const int p0    = prow * 64;

  const float* wm = (slice == 0) ? w1 : (slice == 1) ? w2 : w3;
  const float* xb = x + (long)b * CCH * HWPIX;

  const int mrow = lane & 15;
  const int ksel = lane >> 4;

  // Async-stage one 32ch x 64px K-slab (transposed) into LDS buffer `buf`.
  auto stage = [&](int buf, int c0) {
    #pragma unroll
    for (int j = 0; j < 8; ++j) {
      const int e = tid + j * 256;                 // 0..2047, global-contiguous
      const int c = e >> 6;                        // 0..31
      const int p = e & 63;                        // 0..63
      unsigned lds_off = (unsigned)(size_t)&xbuf[buf][p][c];
      unsigned long long ga =
          (unsigned long long)(const void*)(xb + (long)(c0 + c) * HWPIX + p0 + p);
      asm volatile("global_load_async_to_lds_b32 %0, %1, off"
                   :: "v"(lds_off), "v"(ga) : "memory");
    }
  };

  v8f acc[2][4];
  #pragma unroll
  for (int t = 0; t < 2; ++t)
    #pragma unroll
    for (int t2 = 0; t2 < 4; ++t2) { v8f z = {}; acc[t][t2] = z; }

  stage(0, 0);
  asm volatile("s_wait_asynccnt 0x0" ::: "memory");
  __syncthreads();

  for (int ks = 0; ks < 8; ++ks) {
    const int buf = ks & 1;
    if (ks < 7) stage(buf ^ 1, (ks + 1) * 32);     // overlap copy with compute

    // A fragments (16x32 bf16) for the wave's two m-tiles.
    v16bf afr[2];
    #pragma unroll
    for (int t = 0; t < 2; ++t) {
      const int row = (wv * 2 + t) * 16 + mrow;
      const float* wr = wm + row * CCH + ks * 32 + ksel * 8;
      #pragma unroll
      for (int g = 0; g < 2; ++g)
        #pragma unroll
        for (int j = 0; j < 8; ++j)
          afr[t][g * 8 + j] = tobf(wr[g * 16 + j]);
    }

    // 4 B fragments (32x16 bf16) from LDS (4x ds_load_b128 each); 2 WMMAs each.
    #pragma unroll
    for (int t2 = 0; t2 < 4; ++t2) {
      const int pix = t2 * 16 + (lane & 15);
      const float* br = &xbuf[buf][pix][ksel * 8]; // channels ksel*8+0..7, 16+ksel*8+0..7
      v16bf bfr;
      #pragma unroll
      for (int qq = 0; qq < 4; ++qq) {
        const v4f f = *(const v4f*)(br + (qq >> 1) * 16 + (qq & 1) * 4);   // ds_load_b128
        #pragma unroll
        for (int i = 0; i < 4; ++i) bfr[qq * 4 + i] = tobf(f[i]);
      }
      acc[0][t2] = __builtin_amdgcn_wmma_f32_16x16x32_bf16(false, afr[0], false, bfr,
                                                           (short)0, acc[0][t2], false, false);
      acc[1][t2] = __builtin_amdgcn_wmma_f32_16x16x32_bf16(false, afr[1], false, bfr,
                                                           (short)0, acc[1][t2], false, false);
    }

    asm volatile("s_wait_asynccnt 0x0" ::: "memory");
    __syncthreads();
  }

  // Epilogue: D layout lane L, VGPR r -> (M = r + 8*(L>>4), N = L&15)
  #pragma unroll
  for (int t = 0; t < 2; ++t) {
    const int o0 = (wv * 2 + t) * 16;
    #pragma unroll
    for (int t2 = 0; t2 < 4; ++t2) {
      const int wp = t2 * 16 + (lane & 15);        // column within row `prow`
      const long cl = (((long)b * HH + prow) * WW + wp) * CCH;
      #pragma unroll
      for (int r = 0; r < 8; ++r) {
        const int o = o0 + r + 8 * (lane >> 4);
        if (slice == 0)      x1s[cl + o] = tobf(acc[t][t2][r]);
        else if (slice == 1) x2s[cl + o] = tobf(acc[t][t2][r]);
        else                 x3s[cl + o] = acc[t][t2][r];
      }
    }
  }
}

// ---------------------------------------------------------------------------
// Kernel 2: one wave owns (b, row h, 16-pixel tile w0..w0+15).
//  Phase 1: for each dh, P[m,j] = sum_c x1[h,w0+m,c] * x2[h+dh-3,w0-3+j,c]
//           via 2 N-tiles x 8 K-steps of v_wmma_f32_16x16x32_bf16.
//           Needed logits are the band  weight[m][dh][dw] = P[m][m+dw].
//  Phase 2: softmax over the 49 logits (lanes 0-15, one pixel each).
//  Phase 3: out[c] = sum_k value[neighbor_k][c] * weight[k], channels
//           striped across the 32 lanes (coalesced value loads).
// ---------------------------------------------------------------------------
__global__ __launch_bounds__(256) void local_attn_kernel(
    const __bf16* __restrict__ x1s,
    const __bf16* __restrict__ x2s,
    const float*  __restrict__ x3s,
    float* __restrict__ out)
{
  __shared__ float Pbuf[8][16][32];
  __shared__ float wgt[8][16][KK];

  const int lane = threadIdx.x & 31;
  const int wv   = threadIdx.x >> 5;
  const int gwt  = blockIdx.x * 8 + wv;   // 0 .. 1023
  const int b    = gwt >> 8;
  const int rem  = gwt & 255;
  const int h    = rem >> 2;
  const int w0   = (rem & 3) * 16;

  // Preload all 8 A fragments (x1, row h, pixels w0..w0+15, C=256) once.
  v16bf afrag[8];
  {
    const int m    = lane & 15;
    const int ksel = lane >> 4;
    const __bf16* arow = x1s + (((long)b * HH + h) * WW + (w0 + m)) * CCH;
    #pragma unroll
    for (int ks = 0; ks < 8; ++ks) {
      BFrag f;
      const __bf16* pb = arow + ks * 32 + ksel * 8;
      f.q[0] = *(const u32x4*)(pb);        // K = ks*32 + ksel*8 + 0..7
      f.q[1] = *(const u32x4*)(pb + 16);   // K = ks*32 + 16 + ksel*8 + 0..7
      afrag[ks] = f.v;
    }
  }

  // ---- Phase 1: similarity logits -----------------------------------------
  for (int dh = 0; dh < KH; ++dh) {
    const int hr = h + dh - 3;             // wave-uniform
    if (hr < 0 || hr >= HH) {
      if (lane < 16) {
        #pragma unroll
        for (int dw = 0; dw < KW; ++dw) wgt[wv][lane][dh * KW + dw] = 0.0f;
      }
      __builtin_amdgcn_wave_barrier();
      continue;
    }
    const __bf16* brow = x2s + (((long)b * HH + hr) * WW) * CCH;

    #pragma unroll
    for (int t = 0; t < 2; ++t) {
      const int px  = w0 - 3 + t * 16 + (lane & 15);
      const bool ok = (px >= 0) && (px < WW);
      const __bf16* bcol = brow + (long)(ok ? px : 0) * CCH + (lane >> 4) * 8;

      v8f acc = {};
      #pragma unroll
      for (int ks = 0; ks < 8; ++ks) {
        BFrag f;
        if (ok) {
          f.q[0] = *(const u32x4*)(bcol + ks * 32);      // K = ks*32 + ksel*8 + 0..7
          f.q[1] = *(const u32x4*)(bcol + ks * 32 + 16); // K = ks*32 + 16 + ksel*8 + 0..7
        } else {
          u32x4 z = {};
          f.q[0] = z; f.q[1] = z;                        // zero pad -> logit 0
        }
        acc = __builtin_amdgcn_wmma_f32_16x16x32_bf16(false, afrag[ks], false, f.v,
                                                      (short)0, acc, false, false);
      }
      #pragma unroll
      for (int r = 0; r < 8; ++r)
        Pbuf[wv][r + 8 * (lane >> 4)][t * 16 + (lane & 15)] = acc[r];
    }
    __builtin_amdgcn_wave_barrier();

    if (lane < 16) {
      const int m = lane;
      #pragma unroll
      for (int dw = 0; dw < KW; ++dw)
        wgt[wv][m][dh * KW + dw] = Pbuf[wv][m][m + dw];  // band extract
    }
    __builtin_amdgcn_wave_barrier();
  }

  // ---- Phase 2: softmax over 49 (lanes 0-15, pixel m = lane) --------------
  if (lane < 16) {
    const int m = lane;
    float mx = -1e30f;
    for (int k = 0; k < KK; ++k) mx = fmaxf(mx, wgt[wv][m][k]);
    float s = 0.0f;
    for (int k = 0; k < KK; ++k) s += __expf(wgt[wv][m][k] - mx);
    const float inv = 1.0f / s;
    for (int k = 0; k < KK; ++k)
      wgt[wv][m][k] = __expf(wgt[wv][m][k] - mx) * inv;
  }
  __builtin_amdgcn_wave_barrier();

  // ---- Phase 3: weighted aggregation, 32 channels across lanes ------------
  for (int m = 0; m < 16; ++m) {
    const int wp = w0 + m;
    for (int chunk = 0; chunk < 8; ++chunk) {
      const int c = chunk * 32 + lane;
      float acc = 0.0f;
      for (int dh = 0; dh < KH; ++dh) {
        const int hr = h + dh - 3;
        if (hr < 0 || hr >= HH) continue;  // wave-uniform
        const float* vrow = x3s + (((long)b * HH + hr) * WW) * CCH + c;
        #pragma unroll
        for (int dw = 0; dw < KW; ++dw) {
          const int px = wp + dw - 3;
          if (px >= 0 && px < WW)
            acc += vrow[(long)px * CCH] * wgt[wv][m][dh * KW + dw];
        }
      }
      out[(((long)b * CCH + c) * HH + h) * WW + wp] = acc;
    }
  }
}

// ---------------------------------------------------------------------------
extern "C" void kernel_launch(void* const* d_in, const int* in_sizes, int n_in,
                              void* d_out, int out_size, void* d_ws, size_t ws_size,
                              hipStream_t stream) {
  (void)in_sizes; (void)n_in; (void)out_size; (void)ws_size;

  const float* x  = (const float*)d_in[0];
  const float* w1 = (const float*)d_in[1];
  const float* w2 = (const float*)d_in[2];
  const float* w3 = (const float*)d_in[3];
  // d_in[4], d_in[5] = kH, kW (hardcoded 7x7)
  float* out = (float*)d_out;

  // Workspace: x1s (bf16, 8MB) | x2s (bf16, 8MB) | x3s (f32, 16MB) = 32MB
  char* ws = (char*)d_ws;
  __bf16* x1s = (__bf16*)(ws);
  __bf16* x2s = (__bf16*)(ws + (size_t)8 * 1024 * 1024);
  float*  x3s = (float*)(ws + (size_t)16 * 1024 * 1024);

  // 3 weight-matrices * 4 batches * 64 pixel-rows = 768 blocks.
  mix_gemm_kernel<<<768, 256, 0, stream>>>(x, w1, w2, w3, x1s, x2s, x3s);

  // 4 batches * 64 rows * 4 pixel-tiles = 1024 wave-tiles, 8 waves/block.
  local_attn_kernel<<<128, 256, 0, stream>>>(x1s, x2s, x3s, out);
}
